// EncoderLayer_39857296507465
// MI455X (gfx1250) — compile-verified, matrix operand order checked by reading, not run on hardware
//
#include <hip/hip_runtime.h>
#include <hip/hip_bf16.h>
#include <math.h>

typedef unsigned short ushort_t;
typedef unsigned int   uint_t;
typedef __attribute__((ext_vector_type(16))) __bf16 v16bf;
typedef __attribute__((ext_vector_type(8)))  float  v8f;

#define B_   4
#define S_   4096
#define D_   1024
#define H_   16
#define MF_  128
#define DFF_ 4096
#define DH_  64
#define NT_  (B_*S_)

// ---------------- helpers ----------------
__device__ __forceinline__ ushort_t f2bf(float f) {
  union { float f; uint_t u; } v; v.f = f;
  uint_t r = v.u + 0x7FFFu + ((v.u >> 16) & 1u);   // round-to-nearest-even
  return (ushort_t)(r >> 16);
}
__device__ __forceinline__ float bf2f(ushort_t h) {
  union { uint_t u; float f; } v; v.u = ((uint_t)h) << 16;
  return v.f;
}

// ---------------- elementwise casts ----------------
__global__ void cast_f32_bf16(const float* __restrict__ in, ushort_t* __restrict__ out, long long n) {
  long long i = (long long)blockIdx.x * blockDim.x + threadIdx.x;
  long long stride = (long long)gridDim.x * blockDim.x;
  for (; i < n; i += stride) out[i] = f2bf(in[i]);
}

// in [R][C] f32 -> out [C][R] bf16  (weight pre-transpose for async-friendly B tiles)
__global__ void cast_transpose_bf16(const float* __restrict__ in, ushort_t* __restrict__ out,
                                    int R, int C) {
  long long n = (long long)R * C;
  long long i = (long long)blockIdx.x * blockDim.x + threadIdx.x;
  long long stride = (long long)gridDim.x * blockDim.x;
  for (; i < n; i += stride) {
    int r = (int)(i / C), c = (int)(i % C);
    out[(long long)c * R + r] = f2bf(in[i]);
  }
}

// kvF [bh][MF][DH] f32 -> kvT [bh][DH][MF] bf16
__global__ void kv_transpose(const float* __restrict__ kvF, ushort_t* __restrict__ kvT) {
  int i = blockIdx.x * 256 + threadIdx.x;          // 64*128*64
  int bh = i >> 13, rem = i & 8191;
  int m = rem >> 6, d = rem & 63;
  kvT[((long long)bh * DH_ + d) * MF_ + m] = f2bf(kvF[i]);
}

// 0.5*sum(q_head^2) per (b,h,s); q bf16 [NT, D]; out sq[(b*H+h)*S + s]
__global__ void sumsq_half(const ushort_t* __restrict__ q, float* __restrict__ sq) {
  int gid = blockIdx.x * 256 + threadIdx.x;       // 0 .. NT_*H_
  int h  = gid & (H_ - 1);
  int rs = gid >> 4;                              // b*S + s
  int b  = rs >> 12;
  int s  = rs & (S_ - 1);
  const ushort_t* p = q + (long long)rs * D_ + h * DH_;
  float acc = 0.f;
  #pragma unroll 8
  for (int i = 0; i < DH_; i++) { float x = bf2f(p[i]); acc += x * x; }
  sq[((long long)(b * H_ + h)) * S_ + s] = 0.5f * acc;
}

// ksum[bh][m] = sum_s pk[bh][s][m]  (pk bf16 [64][S][MF])
__global__ void ksum_kernel(const ushort_t* __restrict__ pk, float* __restrict__ ksum) {
  int gid = blockIdx.x * 256 + threadIdx.x;       // 0 .. 64*128
  int z = gid >> 7, m = gid & (MF_ - 1);
  const ushort_t* p = pk + (long long)z * S_ * MF_ + m;
  float acc = 0.f;
  for (int s = 0; s < S_; s++) acc += bf2f(p[(long long)s * MF_]);
  ksum[z * MF_ + m] = acc;
}

// z[bh][s] = 1/(dot(pq[bh][s][:], ksum[bh][:]) + eps)
__global__ __launch_bounds__(256) void zden_kernel(const ushort_t* __restrict__ pq,
                                                   const float* __restrict__ ksum,
                                                   float* __restrict__ z) {
  __shared__ float ks[MF_];
  int bh = blockIdx.y;
  if (threadIdx.x < MF_) ks[threadIdx.x] = ksum[bh * MF_ + threadIdx.x];
  __syncthreads();
  int s = blockIdx.x * 256 + threadIdx.x;
  const ushort_t* p = pq + ((long long)bh * S_ + s) * MF_;
  float acc = 0.f;
  #pragma unroll 8
  for (int m = 0; m < MF_; m++) acc += bf2f(p[m]) * ks[m];
  z[(long long)bh * S_ + s] = 1.f / (acc + 1e-6f);
}

// LayerNorm over D=1024: one block per row; writes f32 and optional bf16
__global__ __launch_bounds__(256) void layernorm_kernel(const float* __restrict__ in,
                                                        const float* __restrict__ g,
                                                        const float* __restrict__ be,
                                                        float* __restrict__ outF,
                                                        ushort_t* __restrict__ outB) {
  __shared__ float red[256];
  int row = blockIdx.x, tid = threadIdx.x;
  const float* x = in + (long long)row * D_;
  float lx[4]; float s = 0.f;
  #pragma unroll
  for (int i = 0; i < 4; i++) { lx[i] = x[tid + i * 256]; s += lx[i]; }
  red[tid] = s; __syncthreads();
  for (int o = 128; o > 0; o >>= 1) { if (tid < o) red[tid] += red[tid + o]; __syncthreads(); }
  float mu = red[0] * (1.f / D_);
  __syncthreads();
  float v = 0.f;
  #pragma unroll
  for (int i = 0; i < 4; i++) { float d = lx[i] - mu; v += d * d; }
  red[tid] = v; __syncthreads();
  for (int o = 128; o > 0; o >>= 1) { if (tid < o) red[tid] += red[tid + o]; __syncthreads(); }
  float rstd = rsqrtf(red[0] * (1.f / D_) + 1e-6f);
  #pragma unroll
  for (int i = 0; i < 4; i++) {
    int c = tid + i * 256;
    float y = (lx[i] - mu) * rstd * g[c] + be[c];
    if (outF) outF[(long long)row * D_ + c] = y;
    if (outB) outB[(long long)row * D_ + c] = f2bf(y);
  }
}

// ---------------- WMMA GEMM ----------------
// Non-TRANSA: C = A(bf16,[M][K]) * BT(bf16,[N][K])^T, async-to-LDS double buffered.
// TRANSA:     C = AT(bf16,[K][M])^T * B(bf16,[K][N]), scalar staging (kv = pk^T v).
// Per-z (batch) split strides: off = (z/batchDiv)*s?z1 + (z%batchDiv)*s?z2.
struct GemmParams {
  const ushort_t* A; const ushort_t* B; const float* bias;
  float* Cf; ushort_t* Cb;
  int M, N, K, lda, ldb, ldc;
  long long sAz1, sAz2, sBz1, sBz2, sCz1, sCz2;
  int batchDiv;
  float alpha;
  const float* rowSub;   long long sSub;    // phi: exp(acc - rowSub[row])*alpha + eps
  const float* rowScale; long long sScale;  // attn: acc * rowScale[row]
  const float* residual; int ldres;         // resid: acc + bias + residual[row][col]
  int mode;   // 0 affine->bf16, 2 phi->bf16, 3 plain->f32, 4 rowscale->bf16, 5 elu->bf16, 6 resid->f32
};

union Frag { v16bf v; uint_t u[8]; };

template <bool TRANSA>
__global__ __launch_bounds__(256) void gemm_bf16_kernel(GemmParams p) {
  __shared__ __align__(16) ushort_t As[2][128 * 32];
  __shared__ __align__(16) ushort_t Bs[2][64 * 32];

  const int tid = threadIdx.x;
  const int z  = blockIdx.z;
  const int zb = z / p.batchDiv, zh = z % p.batchDiv;
  const ushort_t* Ag = p.A + (long long)zb * p.sAz1 + (long long)zh * p.sAz2;
  const ushort_t* Bg = p.B + (long long)zb * p.sBz1 + (long long)zh * p.sBz2;
  const long long cOff = (long long)zb * p.sCz1 + (long long)zh * p.sCz2;

  const int tileN = blockIdx.x * 64;
  const int tileM = blockIdx.y * 128;
  const int wid   = tid >> 5;
  const int lane  = tid & 31;
  const int waveM = wid & 3;      // 0..3 -> 32-row slabs
  const int waveN = wid >> 2;     // 0..1 -> 32-col slabs
  const int half  = lane >> 4;
  const int idx   = lane & 15;

  const v8f vzero = {0.f, 0.f, 0.f, 0.f, 0.f, 0.f, 0.f, 0.f};
  v8f acc[2][2];
  acc[0][0] = vzero; acc[0][1] = vzero; acc[1][0] = vzero; acc[1][1] = vzero;

  // fragment gather + 2x2 WMMA from buffer `buf` (ISA lane layout)
  auto compute = [&](int buf) {
    Frag a[2], b[2];
    #pragma unroll
    for (int i = 0; i < 2; i++) {
      int row = waveM * 32 + i * 16 + idx;
      const uint_t* ap = (const uint_t*)&As[buf][row * 32];
      #pragma unroll
      for (int j = 0; j < 4; j++) { a[i].u[j] = ap[half * 4 + j]; a[i].u[4 + j] = ap[8 + half * 4 + j]; }
    }
    #pragma unroll
    for (int i = 0; i < 2; i++) {
      int col = waveN * 32 + i * 16 + idx;
      const uint_t* bp = (const uint_t*)&Bs[buf][col * 32];
      #pragma unroll
      for (int j = 0; j < 4; j++) { b[i].u[j] = bp[half * 4 + j]; b[i].u[4 + j] = bp[8 + half * 4 + j]; }
    }
    #pragma unroll
    for (int i = 0; i < 2; i++)
      #pragma unroll
      for (int j = 0; j < 2; j++)
        acc[i][j] = __builtin_amdgcn_wmma_f32_16x16x32_bf16(
            false, a[i].v, false, b[j].v, (short)0, acc[i][j], false, false);
  };

  if (TRANSA) {
    // ---- scalar-staged path (A transposed in memory, B row-major [K][N]) ----
    for (int k0 = 0; k0 < p.K; k0 += 32) {
      {
        int k = tid >> 3, mc = (tid & 7) * 16;
        const ushort_t* src = Ag + (long long)(k0 + k) * p.lda + tileM + mc;
        ushort_t tmp[16];
        *(uint4*)&tmp[0] = *(const uint4*)src;
        *(uint4*)&tmp[8] = *(const uint4*)(src + 8);
        #pragma unroll
        for (int i = 0; i < 16; i++) As[0][(mc + i) * 32 + k] = tmp[i];
      }
      {
        int n = tid & 63, kc = (tid >> 6) * 8;
        #pragma unroll
        for (int i = 0; i < 8; i++)
          Bs[0][n * 32 + kc + i] = Bg[(long long)(k0 + kc + i) * p.ldb + tileN + n];
      }
      __syncthreads();
      compute(0);
      __syncthreads();
    }
  } else {
    // ---- async-to-LDS double-buffered path (B pre-transposed: BT[n][k], row stride p.ldb) ----
    auto issue_tile = [&](int k0, int buf) {
      // A tile: 128 rows x 64B; each thread DMAs 32B of one row
      int rowA = tid >> 1, offA = (tid & 1) * 32;
      const char* sA = (const char*)(Ag + (long long)(tileM + rowA) * p.lda + k0) + offA;
      uint_t dA = (uint_t)(unsigned long long)(&As[buf][rowA * 32]) + (uint_t)offA;
      asm volatile("global_load_async_to_lds_b128 %0, %1, off"
                   :: "v"(dA), "v"((unsigned long long)sA) : "memory");
      asm volatile("global_load_async_to_lds_b128 %0, %1, off"
                   :: "v"(dA + 16u), "v"((unsigned long long)sA + 16ull) : "memory");
      // B tile: 64 rows x 64B; each thread DMAs 16B of one row
      int nB = tid >> 2, offB = (tid & 3) * 16;
      const char* sB = (const char*)(Bg + (long long)(tileN + nB) * p.ldb + k0) + offB;
      uint_t dB = (uint_t)(unsigned long long)(&Bs[buf][nB * 32]) + (uint_t)offB;
      asm volatile("global_load_async_to_lds_b128 %0, %1, off"
                   :: "v"(dB), "v"((unsigned long long)sB) : "memory");
    };
    const int nK = p.K >> 5;
    issue_tile(0, 0);
    for (int it = 0; it < nK; ++it) {
      int cur = it & 1;
      asm volatile("s_wait_asynccnt 0x0" ::: "memory");   // my DMAs into `cur` done
      __syncthreads();                                     // everyone's done; prev readers of 1-cur done
      if (it + 1 < nK) issue_tile((it + 1) << 5, 1 - cur); // prefetch next into 1-cur
      compute(cur);
    }
  }

  // ---- epilogue ----
  float* Cf = p.Cf ? p.Cf + cOff : nullptr;
  ushort_t* Cb = p.Cb ? p.Cb + cOff : nullptr;
  const float* rowSub   = p.rowSub   ? p.rowSub   + (long long)z * p.sSub   : nullptr;
  const float* rowScale = p.rowScale ? p.rowScale + (long long)z * p.sScale : nullptr;

  #pragma unroll
  for (int i = 0; i < 2; i++) {
    #pragma unroll
    for (int j = 0; j < 2; j++) {
      int rBase = tileM + waveM * 32 + i * 16 + half * 8;
      int col   = tileN + waveN * 32 + j * 16 + idx;
      #pragma unroll
      for (int r = 0; r < 8; r++) {
        int row = rBase + r;
        float v = acc[i][j][r];
        long long ci = (long long)row * p.ldc + col;
        switch (p.mode) {
          case 0: { v = (v + (p.bias ? p.bias[col] : 0.f)) * p.alpha; Cb[ci] = f2bf(v); } break;
          case 2: { v = expf(v - rowSub[row]) * p.alpha + 1e-6f;      Cb[ci] = f2bf(v); } break;
          case 3: { v += (p.bias ? p.bias[col] : 0.f);                Cf[ci] = v;       } break;
          case 4: { v *= rowScale[row];                               Cb[ci] = f2bf(v); } break;
          case 5: { float t = v + (p.bias ? p.bias[col] : 0.f);
                    v = t > 0.f ? t : (expf(t) - 1.f);                Cb[ci] = f2bf(v); } break;
          case 6: { v += p.bias[col] + p.residual[(long long)row * p.ldres + col];
                                                                      Cf[ci] = v;       } break;
        }
      }
    }
  }
}

static void gemm_go(const GemmParams& g, bool transA, int Z, hipStream_t st) {
  dim3 grid((unsigned)(g.N / 64), (unsigned)(g.M / 128), (unsigned)Z);
  if (transA) gemm_bf16_kernel<true><<<grid, 256, 0, st>>>(g);
  else        gemm_bf16_kernel<false><<<grid, 256, 0, st>>>(g);
}

// ---------------- driver ----------------
extern "C" void kernel_launch(void* const* d_in, const int* in_sizes, int n_in,
                              void* d_out, int out_size, void* d_ws, size_t ws_size,
                              hipStream_t stream) {
  const float* x    = (const float*)d_in[0];
  const float* Wq   = (const float*)d_in[1];
  const float* bq   = (const float*)d_in[2];
  const float* Wk   = (const float*)d_in[3];
  const float* bk   = (const float*)d_in[4];
  const float* Wv   = (const float*)d_in[5];
  const float* bv   = (const float*)d_in[6];
  const float* Wo   = (const float*)d_in[7];
  const float* bo   = (const float*)d_in[8];
  const float* omg  = (const float*)d_in[9];
  const float* W1   = (const float*)d_in[10];
  const float* b1   = (const float*)d_in[11];
  const float* W2   = (const float*)d_in[12];
  const float* b2   = (const float*)d_in[13];
  const float* g1   = (const float*)d_in[14];
  const float* be1  = (const float*)d_in[15];
  const float* g2   = (const float*)d_in[16];
  const float* be2  = (const float*)d_in[17];
  (void)in_sizes; (void)n_in; (void)out_size; (void)ws_size;

  char* ws = (char*)d_ws;
  size_t off = 0;
  auto bump = [&](size_t bytes) -> size_t {
    size_t o = off; off += (bytes + 255) & ~(size_t)255; return o;
  };

  const size_t o_xb   = bump((size_t)NT_ * D_ * 2);
  const size_t o_WqT  = bump((size_t)D_ * D_ * 2);
  const size_t o_WkT  = bump((size_t)D_ * D_ * 2);
  const size_t o_WvT  = bump((size_t)D_ * D_ * 2);
  const size_t o_WoT  = bump((size_t)D_ * D_ * 2);
  const size_t o_W1T  = bump((size_t)D_ * DFF_ * 2);
  const size_t o_W2T  = bump((size_t)DFF_ * D_ * 2);
  const size_t o_omB  = bump((size_t)MF_ * DH_ * 2);
  const size_t o_pool = off;                       // reused later for FFN hidden
  const size_t o_qb   = bump((size_t)NT_ * D_ * 2);
  const size_t o_kb   = bump((size_t)NT_ * D_ * 2);
  const size_t o_vb   = bump((size_t)NT_ * D_ * 2);
  const size_t o_sqq  = bump((size_t)B_ * H_ * S_ * 4);
  const size_t o_sqk  = bump((size_t)B_ * H_ * S_ * 4);
  const size_t o_pq   = bump((size_t)B_ * H_ * S_ * MF_ * 2);
  const size_t o_pk   = bump((size_t)B_ * H_ * S_ * MF_ * 2);
  const size_t o_ksum = bump((size_t)B_ * H_ * MF_ * 4);
  const size_t o_zbuf = bump((size_t)B_ * H_ * S_ * 4);
  const size_t o_kvF  = bump((size_t)B_ * H_ * MF_ * DH_ * 4);
  const size_t o_kvT  = bump((size_t)B_ * H_ * MF_ * DH_ * 2);
  const size_t o_attn = bump((size_t)NT_ * D_ * 2);
  const size_t o_y1   = bump((size_t)NT_ * D_ * 4);
  const size_t o_out1 = bump((size_t)NT_ * D_ * 4);
  const size_t o_o1B  = bump((size_t)NT_ * D_ * 2);
  const size_t o_ffh  = o_pool;   // alias: q/k/v/sq/pq pool is dead by FFN time
  const size_t o_y2   = o_y1;     // alias: y1 dead after LN1

  ushort_t* xb   = (ushort_t*)(ws + o_xb);
  ushort_t* WqT  = (ushort_t*)(ws + o_WqT);
  ushort_t* WkT  = (ushort_t*)(ws + o_WkT);
  ushort_t* WvT  = (ushort_t*)(ws + o_WvT);
  ushort_t* WoT  = (ushort_t*)(ws + o_WoT);
  ushort_t* W1T  = (ushort_t*)(ws + o_W1T);
  ushort_t* W2T  = (ushort_t*)(ws + o_W2T);
  ushort_t* omB  = (ushort_t*)(ws + o_omB);
  ushort_t* qb   = (ushort_t*)(ws + o_qb);
  ushort_t* kb   = (ushort_t*)(ws + o_kb);
  ushort_t* vb   = (ushort_t*)(ws + o_vb);
  float*    sqq  = (float*)(ws + o_sqq);
  float*    sqk  = (float*)(ws + o_sqk);
  ushort_t* pq   = (ushort_t*)(ws + o_pq);
  ushort_t* pk   = (ushort_t*)(ws + o_pk);
  float*    ksum = (float*)(ws + o_ksum);
  float*    zbuf = (float*)(ws + o_zbuf);
  float*    kvF  = (float*)(ws + o_kvF);
  ushort_t* kvT  = (ushort_t*)(ws + o_kvT);
  ushort_t* attnB= (ushort_t*)(ws + o_attn);
  float*    y1   = (float*)(ws + o_y1);
  float*    out1 = (float*)(ws + o_out1);
  ushort_t* o1B  = (ushort_t*)(ws + o_o1B);
  ushort_t* ffhB = (ushort_t*)(ws + o_ffh);
  float*    y2   = (float*)(ws + o_y2);

  const float scaleQK  = 0.35355339059327373f;  // 64^-0.25
  const float invSqrtM = 0.08838834764831845f;  // 1/sqrt(128)

  // casts / pre-transposes to bf16
  cast_f32_bf16<<<4096, 256, 0, stream>>>(x, xb, (long long)NT_ * D_);
  cast_transpose_bf16<<<1024, 256, 0, stream>>>(Wq, WqT, D_, D_);
  cast_transpose_bf16<<<1024, 256, 0, stream>>>(Wk, WkT, D_, D_);
  cast_transpose_bf16<<<1024, 256, 0, stream>>>(Wv, WvT, D_, D_);
  cast_transpose_bf16<<<1024, 256, 0, stream>>>(Wo, WoT, D_, D_);
  cast_transpose_bf16<<<2048, 256, 0, stream>>>(W1, W1T, D_, DFF_);
  cast_transpose_bf16<<<2048, 256, 0, stream>>>(W2, W2T, DFF_, D_);
  cast_f32_bf16<<<32, 256, 0, stream>>>(omg, omB, (long long)MF_ * DH_);  // BT for phi == omega itself

  // q/k/v projections
  { GemmParams g{}; g.A=xb; g.B=WqT; g.bias=bq; g.Cb=qb; g.M=NT_; g.N=D_; g.K=D_;
    g.lda=D_; g.ldb=D_; g.ldc=D_; g.batchDiv=1; g.alpha=scaleQK; g.mode=0;
    gemm_go(g, false, 1, stream); }
  { GemmParams g{}; g.A=xb; g.B=WkT; g.bias=bk; g.Cb=kb; g.M=NT_; g.N=D_; g.K=D_;
    g.lda=D_; g.ldb=D_; g.ldc=D_; g.batchDiv=1; g.alpha=scaleQK; g.mode=0;
    gemm_go(g, false, 1, stream); }
  { GemmParams g{}; g.A=xb; g.B=WvT; g.bias=bv; g.Cb=vb; g.M=NT_; g.N=D_; g.K=D_;
    g.lda=D_; g.ldb=D_; g.ldc=D_; g.batchDiv=1; g.alpha=1.0f; g.mode=0;
    gemm_go(g, false, 1, stream); }

  // 0.5*|u|^2 per head-row
  sumsq_half<<<NT_ * H_ / 256, 256, 0, stream>>>(qb, sqq);
  sumsq_half<<<NT_ * H_ / 256, 256, 0, stream>>>(kb, sqk);

  // phi(q), phi(k): per (b,h) GEMM [S,DH]x[DH,MF] with exp epilogue; BT = omega [MF][DH]
  { GemmParams g{}; g.A=qb; g.B=omB; g.Cb=pq; g.M=S_; g.N=MF_; g.K=DH_;
    g.lda=D_; g.ldb=DH_; g.ldc=MF_; g.batchDiv=H_;
    g.sAz1=(long long)S_*D_; g.sAz2=DH_;
    g.sCz1=(long long)H_*S_*MF_; g.sCz2=(long long)S_*MF_;
    g.alpha=invSqrtM; g.rowSub=sqq; g.sSub=S_; g.mode=2;
    gemm_go(g, false, B_*H_, stream); }
  { GemmParams g{}; g.A=kb; g.B=omB; g.Cb=pk; g.M=S_; g.N=MF_; g.K=DH_;
    g.lda=D_; g.ldb=DH_; g.ldc=MF_; g.batchDiv=H_;
    g.sAz1=(long long)S_*D_; g.sAz2=DH_;
    g.sCz1=(long long)H_*S_*MF_; g.sCz2=(long long)S_*MF_;
    g.alpha=invSqrtM; g.rowSub=sqk; g.sSub=S_; g.mode=2;
    gemm_go(g, false, B_*H_, stream); }

  // ksum over S
  ksum_kernel<<<B_ * H_ * MF_ / 256, 256, 0, stream>>>(pk, ksum);

  // kv = pk^T @ v per (b,h): TRANS_A scalar path (pk stored [S][MF], v is [S][D] slice)
  { GemmParams g{}; g.A=pk; g.B=vb; g.Cf=kvF; g.M=MF_; g.N=DH_; g.K=S_;
    g.lda=MF_; g.ldb=D_; g.ldc=DH_; g.batchDiv=H_;
    g.sAz1=(long long)H_*S_*MF_; g.sAz2=(long long)S_*MF_;
    g.sBz1=(long long)S_*D_; g.sBz2=DH_;
    g.sCz1=(long long)H_*MF_*DH_; g.sCz2=(long long)MF_*DH_;
    g.mode=3;
    gemm_go(g, true, B_*H_, stream); }
  kv_transpose<<<B_ * H_ * MF_ * DH_ / 256, 256, 0, stream>>>(kvF, kvT);

  // z = 1/(pq . ksum + eps)
  zden_kernel<<<dim3(S_ / 256, B_ * H_), 256, 0, stream>>>(pq, ksum, zbuf);

  // attn = (pq @ kv) * z, merged-head bf16 store (ldc = D); BT = kvT [DH][MF]
  { GemmParams g{}; g.A=pq; g.B=kvT; g.Cb=attnB; g.M=S_; g.N=DH_; g.K=MF_;
    g.lda=MF_; g.ldb=MF_; g.ldc=D_; g.batchDiv=H_;
    g.sAz1=(long long)H_*S_*MF_; g.sAz2=(long long)S_*MF_;
    g.sBz1=(long long)H_*DH_*MF_; g.sBz2=(long long)DH_*MF_;
    g.sCz1=(long long)S_*D_; g.sCz2=DH_;
    g.rowScale=zbuf; g.sScale=S_; g.mode=4;
    gemm_go(g, false, B_*H_, stream); }

  // y1 = attn @ Wo + bo + x  (fused residual)
  { GemmParams g{}; g.A=attnB; g.B=WoT; g.bias=bo; g.Cf=y1; g.M=NT_; g.N=D_; g.K=D_;
    g.lda=D_; g.ldb=D_; g.ldc=D_; g.batchDiv=1;
    g.residual=x; g.ldres=D_; g.mode=6;
    gemm_go(g, false, 1, stream); }

  // out1 = LN(y1)
  layernorm_kernel<<<NT_, 256, 0, stream>>>(y1, g1, be1, out1, o1B);

  // ffh = elu(out1 @ W1 + b1)
  { GemmParams g{}; g.A=o1B; g.B=W1T; g.bias=b1; g.Cb=ffhB; g.M=NT_; g.N=DFF_; g.K=D_;
    g.lda=D_; g.ldb=D_; g.ldc=DFF_; g.batchDiv=1; g.mode=5;
    gemm_go(g, false, 1, stream); }

  // y2 = ffh @ W2 + b2 + out1  (fused residual)
  { GemmParams g{}; g.A=ffhB; g.B=W2T; g.bias=b2; g.Cf=y2; g.M=NT_; g.N=D_; g.K=DFF_;
    g.lda=DFF_; g.ldb=DFF_; g.ldc=D_; g.batchDiv=1;
    g.residual=out1; g.ldres=D_; g.mode=6;
    gemm_go(g, false, 1, stream); }

  // out2 = LN(y2) -> d_out (f32)
  layernorm_kernel<<<NT_, 256, 0, stream>>>(y2, g2, be2, (float*)d_out, nullptr);
}
